// GAT_LSTM_model_83150566851131
// MI455X (gfx1250) — compile-verified, matrix operand order checked by reading, not run on hardware
//
#include <hip/hip_runtime.h>
#include <hip/hip_bf16.h>
#include <cstddef>

// ---------------- problem constants ----------------
#define BS        1024
#define NUM_PATCH 128
#define PATCH     128
#define FEAT      11
#define HID       64
#define LSTM_H    64

typedef __attribute__((ext_vector_type(16))) _Float16 v16h;
typedef __attribute__((ext_vector_type(8)))  _Float16 v8h;
typedef __attribute__((ext_vector_type(8)))  float    v8f;

__device__ __forceinline__ float lrelu(float x, float a) { return x > 0.f ? x : a * x; }
// one v_exp_f32 each; avoids the branchy OCML tanhf/expf paths
__device__ __forceinline__ float fsigm(float x) {
  return __fdividef(1.f, 1.f + __expf(-x));
}
__device__ __forceinline__ float ftanh(float x) {
  float e = __expf(-2.f * x);
  return __fdividef(1.f - e, 1.f + e);
}

// =====================================================================
// Kernel 1: statistical feature extraction.
// 1 wave per 128-sample patch, one float4 per lane in registers,
// wave32 butterfly reductions (no LDS, no barriers). HBM-bound (~3us floor
// for the 64MB read at 23.3 TB/s).
// grid = (BS*NUM_PATCH)/4 blocks of 128 threads.
// =====================================================================
__global__ __launch_bounds__(128) void feat_kernel(const float* __restrict__ x,
                                                   float* __restrict__ feats) {
  const int wave = threadIdx.x >> 5;
  const int lane = threadIdx.x & 31;
  const int r = blockIdx.x * 4 + wave;                 // patch row
  const float4 vv = ((const float4*)(x + (size_t)r * PATCH))[lane];
  float v[4] = {vv.x, vv.y, vv.z, vv.w};

  float s1 = 0.f, s2 = 0.f, sab = 0.f, ssq = 0.f;
  float mx = -3.4e38f, mn = 3.4e38f;
#pragma unroll
  for (int q = 0; q < 4; ++q) {
    float t = v[q];
    s1 += t; s2 += t * t;
    float a = fabsf(t);
    sab += a; ssq += sqrtf(a);
    mx = fmaxf(mx, t); mn = fminf(mn, t);
  }
#pragma unroll
  for (int m = 16; m > 0; m >>= 1) {
    s1 += __shfl_xor(s1, m, 32);
    s2 += __shfl_xor(s2, m, 32);
    sab += __shfl_xor(sab, m, 32);
    ssq += __shfl_xor(ssq, m, 32);
    mx = fmaxf(mx, __shfl_xor(mx, m, 32));
    mn = fminf(mn, __shfl_xor(mn, m, 32));
  }
  const float M = (float)PATCH;
  float mean = s1 * (1.f / M);

  float s3 = 0.f, s4 = 0.f;
#pragma unroll
  for (int q = 0; q < 4; ++q) {
    float md = v[q] - mean;
    float md2 = md * md;
    s3 += md2 * md;
    s4 += md2 * md2;
  }
#pragma unroll
  for (int m = 16; m > 0; m >>= 1) {
    s3 += __shfl_xor(s3, m, 32);
    s4 += __shfl_xor(s4, m, 32);
  }

  if (lane == 0) {
    float var = (s2 - M * mean * mean) / (M - 1.f);     // unbiased
    float stdv = sqrtf(var);
    float rms_amp = ssq / M; rms_amp *= rms_amp;
    float rms = sqrtf(s2 / M);
    float ptp = 0.5f * (mx - mn);
    float c_sk = M / ((M - 1.f) * (M - 2.f));
    float skew = c_sk * s3 / (stdv * stdv * stdv);
    float mm1 = M - 1.f;
    float c_ku = (M * (M + 1.f) - 3.f * mm1 * mm1 * mm1) /
                 ((M - 1.f) * (M - 2.f) * (M - 3.f));
    float kurt = c_ku * s4 / (var * var);
    float absmax = fmaxf(fabsf(mx), fabsf(mn));
    float absmean = sab / M;
    float* o = feats + (size_t)r * FEAT;
    o[0] = mean;  o[1] = stdv;  o[2] = rms_amp; o[3] = rms; o[4] = ptp;
    o[5] = skew;  o[6] = kurt;
    o[7] = absmax / rms;       // crest
    o[8] = absmax / rms_amp;   // clearance
    o[9] = rms / absmean;      // shape
    o[10] = absmax / absmean;  // impulse
  }
}

// =====================================================================
// Kernel 2: fused GAT layer 0 + layer 1, one batch sample per block.
// Exploits: (a) tridiagonal mask => att@Wh needs only 3 neighbors;
// (b) leaky-relu monotone => softmax row-max = lrelu(f1_i + max_j f2_j + ab).
// Everything (Wh, h1, weights) stays in ~96KB LDS (320KB/WGP on CDNA5).
// grid = BS blocks of 256 threads.
// =====================================================================
__global__ __launch_bounds__(256) void gat2_kernel(
    const float* __restrict__ feats,
    const float* __restrict__ W0, const float* __restrict__ b0,
    const float* __restrict__ a0, const float* __restrict__ ab0,
    const float* __restrict__ W1, const float* __restrict__ b1,
    const float* __restrict__ a1, const float* __restrict__ ab1,
    float* __restrict__ h2) {
  __shared__ float fs[NUM_PATCH][FEAT + 1];
  __shared__ float Whs[NUM_PATCH][HID + 1];
  __shared__ float h1s[NUM_PATCH][HID + 1];
  __shared__ float W0s[HID * FEAT];
  __shared__ float W1s[HID * HID];
  __shared__ float b0s[HID], b1s[HID];
  __shared__ float a0s[2 * HID], a1s[2 * HID];
  __shared__ float f1s[NUM_PATCH], f2s[NUM_PATCH];
  __shared__ float wcoef[NUM_PATCH][3];
  __shared__ float f2max;

  const int tid = threadIdx.x;
  const int b = blockIdx.x;

  for (int i = tid; i < HID * FEAT; i += 256) W0s[i] = W0[i];
  for (int i = tid; i < HID * HID; i += 256) W1s[i] = W1[i];
  if (tid < HID) { b0s[tid] = b0[tid]; b1s[tid] = b1[tid]; }
  if (tid < 2 * HID) { a0s[tid] = a0[tid]; a1s[tid] = a1[tid]; }
  for (int idx = tid; idx < NUM_PATCH * FEAT; idx += 256) {
    int i = idx / FEAT, f = idx % FEAT;
    fs[i][f] = feats[((size_t)b * NUM_PATCH + i) * FEAT + f];
  }
  __syncthreads();

  // ---- layer 0 projection: Wh = feats @ W0^T + b0 ----
  for (int idx = tid; idx < NUM_PATCH * HID; idx += 256) {
    int i = idx >> 6, dd = idx & 63;
    float s = b0s[dd];
#pragma unroll
    for (int f = 0; f < FEAT; ++f) s += fs[i][f] * W0s[dd * FEAT + f];
    Whs[i][dd] = s;
  }
  __syncthreads();

  for (int layer = 0; layer < 2; ++layer) {
    const float* aw = layer ? a1s : a0s;
    float abv = layer ? ab1[0] : ab0[0];

    if (tid < NUM_PATCH) {
      float s1 = 0.f, s2 = 0.f;
#pragma unroll 8
      for (int dd = 0; dd < HID; ++dd) {
        float w = Whs[tid][dd];
        s1 += w * aw[dd];
        s2 += w * aw[HID + dd];
      }
      f1s[tid] = s1; f2s[tid] = s2;
    }
    __syncthreads();
    if (tid == 0) {
      float m = f2s[0];
      for (int j = 1; j < NUM_PATCH; ++j) m = fmaxf(m, f2s[j]);
      f2max = m;
    }
    __syncthreads();
    if (tid < NUM_PATCH) {
      int i = tid;
      float mi = lrelu(f1s[i] + f2max + abv, 0.1f);   // exact row max (monotone)
      float den = 0.f;
      for (int j = 0; j < NUM_PATCH; ++j)
        den += __expf(lrelu(f1s[i] + f2s[j] + abv, 0.1f) - mi);
      float inv = __fdividef(1.f, den);
#pragma unroll
      for (int kk = 0; kk < 3; ++kk) {
        int j = i - 1 + kk;
        float w = 0.f;
        if (j >= 0 && j < NUM_PATCH)
          w = __expf(lrelu(f1s[i] + f2s[j] + abv, 0.1f) - mi) * inv;
        wcoef[i][kk] = w;
      }
    }
    __syncthreads();

    if (layer == 0) {
      // h1 = lrelu(att @ Wh, 0.01)
      for (int idx = tid; idx < NUM_PATCH * HID; idx += 256) {
        int i = idx >> 6, dd = idx & 63;
        float s = 0.f;
#pragma unroll
        for (int kk = 0; kk < 3; ++kk) {
          int j = i - 1 + kk;
          if (j >= 0 && j < NUM_PATCH) s += wcoef[i][kk] * Whs[j][dd];
        }
        h1s[i][dd] = lrelu(s, 0.01f);
      }
      __syncthreads();
      // layer 1 projection: Wh = h1 @ W1^T + b1
      for (int idx = tid; idx < NUM_PATCH * HID; idx += 256) {
        int i = idx >> 6, dd = idx & 63;
        float s = b1s[dd];
#pragma unroll 8
        for (int k = 0; k < HID; ++k) s += h1s[i][k] * W1s[dd * HID + k];
        Whs[i][dd] = s;
      }
      __syncthreads();
    } else {
      for (int idx = tid; idx < NUM_PATCH * HID; idx += 256) {
        int i = idx >> 6, dd = idx & 63;
        float s = 0.f;
#pragma unroll
        for (int kk = 0; kk < 3; ++kk) {
          int j = i - 1 + kk;
          if (j >= 0 && j < NUM_PATCH) s += wcoef[i][kk] * Whs[j][dd];
        }
        h2[((size_t)b * NUM_PATCH + i) * HID + dd] = lrelu(s, 0.01f);
      }
    }
  }
}

// =====================================================================
// Kernel 3: LSTM with WMMA (v_wmma_f32_16x16x32_f16).
// Block = 128 threads (4 waves), 16-row batch tile over 128 steps.
// Wave w owns hidden dims [16w,16w+16) for ALL four gates, so the c/h
// elementwise update stays in the wave's D-fragment layout (no shuffles).
// Weights live in B-fragments in registers. x_{t+1} is software-pipelined:
// vector loads (2x b128/thread) issued before the step-t WMMAs, landed into
// a double-buffered LDS tile after them, hiding the full global latency
// behind the 16 WMMAs.
// grid = BS/16 = 64 blocks.
// =====================================================================
__global__ __launch_bounds__(128) void lstm_kernel(
    const float* __restrict__ xin,   // (BS*128, 64)  GAT output
    const float* __restrict__ Wih,   // (256, 64)
    const float* __restrict__ Whh,   // (256, 64)
    const float* __restrict__ bih, const float* __restrict__ bhh,
    float* __restrict__ ys) {        // (BS, 128, 64)
  __shared__ _Float16 xs[2][16][64];
  __shared__ _Float16 hs[16][64];

  const int tid = threadIdx.x;
  const int wave = tid >> 5;          // 0..3
  const int lane = tid & 31;
  const int ln16 = lane & 15;         // A: row m ; B: col n ; D: col n
  const int hi = lane >> 4;           // K-half select / D row offset
  const int nbase = wave * 16;        // hidden dims owned by this wave
  const int bbase = blockIdx.x * 16;  // batch tile

  // staging map: thread handles 8 contiguous elements of row sm
  const int sm = tid >> 3;            // 0..15
  const int sd = (tid & 7) * 8;       // 0,8,...,56

  // ---- load B fragments (weights as f16), once ----
  v16h Bf[4][2][2];                   // [gate][mat(ih,hh)][kchunk]
#pragma unroll
  for (int g = 0; g < 4; ++g) {
    int c = g * 64 + nbase + ln16;    // weight row == gate column
#pragma unroll
    for (int kc = 0; kc < 2; ++kc) {
#pragma unroll
      for (int e = 0; e < 16; ++e) {
        int ko = kc * 32 + ((e >> 3) << 4) + 8 * hi + (e & 7);
        Bf[g][0][kc][e] = (_Float16)Wih[c * 64 + ko];
        Bf[g][1][kc][e] = (_Float16)Whh[c * 64 + ko];
      }
    }
  }
  float bias[4];
#pragma unroll
  for (int g = 0; g < 4; ++g) {
    int c = g * 64 + nbase + ln16;
    bias[g] = bih[c] + bhh[c];
  }

  // prologue: stage x_0, zero h_0
  {
    const float4* src =
        (const float4*)(xin + (((size_t)(bbase + sm)) * NUM_PATCH + 0) * 64 + sd);
    float4 q0 = src[0], q1 = src[1];
    v8h p = {(_Float16)q0.x, (_Float16)q0.y, (_Float16)q0.z, (_Float16)q0.w,
             (_Float16)q1.x, (_Float16)q1.y, (_Float16)q1.z, (_Float16)q1.w};
    *(v8h*)&xs[0][sm][sd] = p;
    v8h z = {};
    *(v8h*)&hs[sm][sd] = z;
  }
  __syncthreads();

  v8f cst = {};                       // cell state in D layout

  for (int t = 0; t < NUM_PATCH; ++t) {
    const int cur = t & 1, nxt = cur ^ 1;

    // issue next step's global loads early (consumed after the WMMAs)
    float4 q0, q1;
    if (t + 1 < NUM_PATCH) {
      const float4* src =
          (const float4*)(xin + (((size_t)(bbase + sm)) * NUM_PATCH + (t + 1)) * 64 + sd);
      q0 = src[0]; q1 = src[1];
    }

    // read A fragments: h_{t-1} and x_t
    v16h Ah[2], Ax[2];
#pragma unroll
    for (int kc = 0; kc < 2; ++kc)
#pragma unroll
      for (int e = 0; e < 16; ++e) {
        int ko = kc * 32 + ((e >> 3) << 4) + 8 * hi + (e & 7);
        Ah[kc][e] = hs[ln16][ko];
        Ax[kc][e] = xs[cur][ln16][ko];
      }
    __syncthreads();  // all hs / xs[cur] reads complete

    v8f acc[4];
#pragma unroll
    for (int g = 0; g < 4; ++g) {
      v8f a = {};
#pragma unroll
      for (int kc = 0; kc < 2; ++kc)
        a = __builtin_amdgcn_wmma_f32_16x16x32_f16(false, Ax[kc], false,
                                                   Bf[g][0][kc], (short)0, a,
                                                   false, false);
#pragma unroll
      for (int kc = 0; kc < 2; ++kc)
        a = __builtin_amdgcn_wmma_f32_16x16x32_f16(false, Ah[kc], false,
                                                   Bf[g][1][kc], (short)0, a,
                                                   false, false);
      acc[g] = a;
    }

    // land the prefetched x_{t+1} (loads have had the WMMA phase to complete)
    if (t + 1 < NUM_PATCH) {
      v8h p = {(_Float16)q0.x, (_Float16)q0.y, (_Float16)q0.z, (_Float16)q0.w,
               (_Float16)q1.x, (_Float16)q1.y, (_Float16)q1.z, (_Float16)q1.w};
      *(v8h*)&xs[nxt][sm][sd] = p;
    }

    // elementwise gate math in D layout: elem v -> (row m=v+8*hi, col ln16)
#pragma unroll
    for (int v = 0; v < 8; ++v) {
      float gi = fsigm(acc[0][v] + bias[0]);
      float gf = fsigm(acc[1][v] + bias[1]);
      float gg = ftanh(acc[2][v] + bias[2]);
      float go = fsigm(acc[3][v] + bias[3]);
      float c = gf * cst[v] + gi * gg;
      cst[v] = c;
      float h = go * ftanh(c);
      int m = v + 8 * hi;
      hs[m][nbase + ln16] = (_Float16)h;
      ys[(((size_t)(bbase + m)) * NUM_PATCH + t) * 64 + nbase + ln16] = h;
    }
    __syncthreads();  // h_t / xs[nxt] visible for next iteration
  }
}

// =====================================================================
// Kernel 4: final FC — out[b] = ys[b,:] . fcW + fcb. One block per sample,
// float4 loads (33.5MB read, HBM-bound).
// =====================================================================
__global__ __launch_bounds__(256) void fc_kernel(const float* __restrict__ ys,
                                                 const float* __restrict__ fcW,
                                                 const float* __restrict__ fcb,
                                                 float* __restrict__ out) {
  __shared__ float red[256];
  const int b = blockIdx.x;
  const float4* ya = (const float4*)(ys + (size_t)b * (NUM_PATCH * LSTM_H));
  const float4* wa = (const float4*)fcW;
  float s = 0.f;
  for (int k = threadIdx.x; k < (NUM_PATCH * LSTM_H) / 4; k += 256) {
    float4 a = ya[k], w = wa[k];
    s += a.x * w.x + a.y * w.y + a.z * w.z + a.w * w.w;
  }
  red[threadIdx.x] = s;
  __syncthreads();
  for (int off = 128; off > 0; off >>= 1) {
    if (threadIdx.x < off) red[threadIdx.x] += red[threadIdx.x + off];
    __syncthreads();
  }
  if (threadIdx.x == 0) out[b] = red[0] + fcb[0];
}

// =====================================================================
extern "C" void kernel_launch(void* const* d_in, const int* in_sizes, int n_in,
                              void* d_out, int out_size, void* d_ws, size_t ws_size,
                              hipStream_t stream) {
  const float* x   = (const float*)d_in[0];
  const float* W0  = (const float*)d_in[1];
  const float* b0  = (const float*)d_in[2];
  const float* a0  = (const float*)d_in[3];
  const float* ab0 = (const float*)d_in[4];
  const float* W1  = (const float*)d_in[5];
  const float* b1  = (const float*)d_in[6];
  const float* a1  = (const float*)d_in[7];
  const float* ab1 = (const float*)d_in[8];
  const float* Wih = (const float*)d_in[9];
  const float* Whh = (const float*)d_in[10];
  const float* bih = (const float*)d_in[11];
  const float* bhh = (const float*)d_in[12];
  const float* fcW = (const float*)d_in[13];
  const float* fcb = (const float*)d_in[14];
  float* out = (float*)d_out;

  const size_t ROWS = (size_t)BS * NUM_PATCH;   // 131072
  float* feats = (float*)d_ws;                  // ROWS * 11
  float* h2 = feats + ROWS * FEAT;              // ROWS * 64
  float* ys = h2 + ROWS * HID;                  // ROWS * 64
  // total workspace: ~73 MB

  feat_kernel<<<ROWS / 4, 128, 0, stream>>>(x, feats);
  gat2_kernel<<<BS, 256, 0, stream>>>(feats, W0, b0, a0, ab0, W1, b1, a1, ab1, h2);
  lstm_kernel<<<BS / 16, 128, 0, stream>>>(h2, Wih, Whh, bih, bhh, ys);
  fc_kernel<<<BS, 256, 0, stream>>>(ys, fcW, fcb, out);
}
